// NN_90245852823746
// MI455X (gfx1250) — matrix operand order repaired and checked by reading, not compiled, not run on hardware
//
#include <hip/hip_runtime.h>

#define NF 128

typedef __attribute__((ext_vector_type(16))) _Float16 v16h;
typedef __attribute__((ext_vector_type(8)))  _Float16 v8h;
typedef __attribute__((ext_vector_type(8)))  float    v8f;

__device__ __forceinline__ float leaky(float x){ return x > 0.0f ? x : 0.2f * x; }

// ---------------- small helper kernels ----------------

__global__ void k_zero_u32(unsigned* __restrict__ p, int n){
  int i = blockIdx.x * 256 + threadIdx.x;
  if (i < n) p[i] = 0u;
}

// set adjacency bits (dedups duplicate edges, matches jnp .at[].set(1))
__global__ void k_adj_bits(const int* __restrict__ conn, unsigned* __restrict__ bits,
                           int E, int words){
  int e = blockIdx.x * 256 + threadIdx.x;
  if (e < E){
    int c0 = conn[2*e], c1 = conn[2*e+1];
    atomicOr(&bits[(size_t)c0 * words + (c1 >> 5)], 1u << (c1 & 31));
    atomicOr(&bits[(size_t)c1 * words + (c0 >> 5)], 1u << (c0 & 31));
  }
}

__global__ void k_degree(const unsigned* __restrict__ bits, int* __restrict__ deg,
                         int Nn, int words){
  int n = blockIdx.x * 256 + threadIdx.x;
  if (n < Nn){
    const unsigned* row = bits + (size_t)n * words;
    int s = 0;
    for (int i = 0; i < words; ++i) s += __popc(row[i]);
    deg[n] = s;
  }
}

// n1[e] = max(deg(c0)-1,1) (self-loop: I1 column empty -> max(-1,1)=1)
// n2[e] = max(deg(c1)-1,1)
__global__ void k_edge_norms(const int* __restrict__ conn, const int* __restrict__ deg,
                             float* __restrict__ inv1, float* __restrict__ inv2, int E){
  int e = blockIdx.x * 256 + threadIdx.x;
  if (e < E){
    int c0 = conn[2*e], c1 = conn[2*e+1];
    int d2 = deg[c1] - 1; if (d2 < 1) d2 = 1;
    inv2[e] = 1.0f / (float)d2;
    if (c0 == c1) inv1[e] = 1.0f;
    else { int d1 = deg[c0] - 1; if (d1 < 1) d1 = 1; inv1[e] = 1.0f / (float)d1; }
  }
}

// hv = leaky(v @ W13.T + b13)   (K=5, trivially small)
__global__ void k_hv(const float* __restrict__ v, const float* __restrict__ W13,
                     const float* __restrict__ b13, float* __restrict__ hv, int n){
  int idx = blockIdx.x * 256 + threadIdx.x;
  if (idx < n){
    int f = idx & (NF-1), nd = idx >> 7;
    float s = b13[f];
    #pragma unroll
    for (int k = 0; k < 5; ++k) s += v[nd*5 + k] * W13[f*5 + k];
    hv[idx] = leaky(s);
  }
}

// h1 = w @ W11.T  (K=4)
__global__ void k_h1(const float* __restrict__ w, const float* __restrict__ W11,
                     float* __restrict__ out, int n){
  int idx = blockIdx.x * 256 + threadIdx.x;
  if (idx < n){
    int f = idx & (NF-1), e = idx >> 7;
    float s = 0.0f;
    #pragma unroll
    for (int k = 0; k < 4; ++k) s += w[e*4 + k] * W11[f*4 + k];
    out[idx] = s;
  }
}

// X0 = IA.T @ hv : hv[c0]+hv[c1] (self-loop: hv[c0] once)
__global__ void k_gatherX0(const int* __restrict__ conn, const float* __restrict__ hv,
                           float* __restrict__ out, int n){
  int idx = blockIdx.x * 256 + threadIdx.x;
  if (idx < n){
    int f = idx & (NF-1), e = idx >> 7;
    int c0 = conn[2*e], c1 = conn[2*e+1];
    float s = hv[(size_t)c0*NF + f];
    if (c1 != c0) s += hv[(size_t)c1*NF + f];
    out[idx] = s;
  }
}

// deterministic scatter IA @ mu via Q32 fixed-point int64 atomics
__global__ void k_scatter(const float* __restrict__ mu, const int* __restrict__ conn,
                          unsigned long long* __restrict__ acc, int n){
  int idx = blockIdx.x * 256 + threadIdx.x;
  if (idx < n){
    int f = idx & (NF-1), e = idx >> 7;
    long long q = (long long)((double)mu[idx] * 4294967296.0);
    int c0 = conn[2*e], c1 = conn[2*e+1];
    atomicAdd(&acc[(size_t)c0*NF + f], (unsigned long long)q);
    if (c1 != c0) atomicAdd(&acc[(size_t)c1*NF + f], (unsigned long long)q);
  }
}

// G1 = I1.T@h4_0 - mu ; G2 = I2.T@h4_0 - mu
__global__ void k_make_G(const float* __restrict__ mu, const int* __restrict__ conn,
                         const unsigned long long* __restrict__ acc,
                         float* __restrict__ G1, float* __restrict__ G2, int n){
  int idx = blockIdx.x * 256 + threadIdx.x;
  if (idx < n){
    int f = idx & (NF-1), e = idx >> 7;
    int c0 = conn[2*e], c1 = conn[2*e+1];
    float m = mu[idx];
    float a1 = (float)((double)(long long)acc[(size_t)c1*NF + f] * (1.0/4294967296.0));
    G2[idx] = a1 - m;
    if (c0 == c1) G1[idx] = -m;  // I1 column empty on self-loop
    else {
      float a0 = (float)((double)(long long)acc[(size_t)c0*NF + f] * (1.0/4294967296.0));
      G1[idx] = a0 - m;
    }
  }
}

// deterministic column sum of mu (one block per feature, fixed tree reduce)
__global__ void k_colsum(const float* __restrict__ mu, float* __restrict__ cs, int E){
  __shared__ float sm[256];
  int f = blockIdx.x, tid = threadIdx.x;
  float s = 0.0f;
  for (int e = tid; e < E; e += 256) s += mu[(size_t)e*NF + f];
  sm[tid] = s; __syncthreads();
  for (int o = 128; o > 0; o >>= 1){
    if (tid < o) sm[tid] += sm[tid + o];
    __syncthreads();
  }
  if (tid == 0) cs[f] = sm[0];
}

__global__ void k_qconst(const float* __restrict__ cs, const float* __restrict__ W21,
                         float* __restrict__ qc){
  for (int a = 0; a < 2; ++a){
    float s = 0.0f;
    for (int f = 0; f < NF; ++f) s += cs[f] * W21[a*(2*NF) + f];
    qc[a] = s;
  }
}

__global__ void k_final(const float* __restrict__ mu, const float* __restrict__ W21,
                        const float* __restrict__ qc, float* __restrict__ out, int E){
  int e = blockIdx.x * 256 + threadIdx.x;
  if (e < E){
    const float* mr = mu + (size_t)e*NF;
    float q0 = qc[0], q1 = qc[1];
    for (int f = 0; f < NF; ++f){
      float m = mr[f];
      q0 += m * W21[NF + f];
      q1 += m * W21[2*NF + NF + f];
    }
    out[2*e]   = q0;
    out[2*e+1] = q1;
  }
}

// ---------------- WMMA GEMM: out[e,0:128] = epi( X[e,:] @ W.T + b ) ----------------
// 256 threads = 8 wave32s; W (128x128) staged in LDS as f16 (stride 136 halves);
// each wave: one 16-row M tile, A-frags in regs, 8 N tiles x 4 K chunks = 32 WMMAs.
// MODE 0: out = leaky(acc + b + addend)
// MODE 1: out = acc + b
// MODE 2: out = leaky((acc + b) * rs[e])
// MODE 3: out = addend + leaky((acc + b) * rs[e])
template<int MODE>
__global__ __launch_bounds__(256)
void k_gemm128_wmma(const float* __restrict__ X, const float* __restrict__ W,
                    const float* __restrict__ bias, const float* __restrict__ addend,
                    const float* __restrict__ rowscale, float* __restrict__ out)
{
  __shared__ __align__(16) _Float16 Wh[NF * 136];
  __shared__ float bsh[NF];
  const int tid = threadIdx.x;

  for (int i = tid; i < NF*NF; i += 256){
    int f = i >> 7, k = i & (NF-1);
    Wh[f*136 + k] = (_Float16)W[i];            // Wh[f][k] = W[f,k] == B[k,f]
  }
  if (tid < NF) bsh[tid] = bias[tid];
  __syncthreads();

  const int wave = tid >> 5;
  const int lane = tid & 31;
  const int hl   = lane >> 4;                  // half-wave select
  const int m    = lane & 15;
  const int rowBase = blockIdx.x * 128 + wave * 16;
  const int kbA = hl ? 8  : 0;                 // A frag: K groups {kbA..kbA+7, kbA+16..kbA+23}
  const int kbB = hl ? 8  : 0;                 // B frag: K groups {kbB..kbB+7, kbB+16..kbB+23}

  // A fragments (16x32 f16 layout), 4 chunks of K=32
  v16h a[4];
  const float4* xr = (const float4*)(X + (size_t)(rowBase + m) * NF);
  #pragma unroll
  for (int kc = 0; kc < 4; ++kc){
    int b0 = (kc*32 + kbA) >> 2;
    float4 p0 = xr[b0], p1 = xr[b0+1];         // K = kbA..kbA+7
    float4 p2 = xr[b0+4], p3 = xr[b0+5];       // K = kbA+16..kbA+23
    a[kc][0]=(_Float16)p0.x; a[kc][1]=(_Float16)p0.y; a[kc][2]=(_Float16)p0.z; a[kc][3]=(_Float16)p0.w;
    a[kc][4]=(_Float16)p1.x; a[kc][5]=(_Float16)p1.y; a[kc][6]=(_Float16)p1.z; a[kc][7]=(_Float16)p1.w;
    a[kc][8]=(_Float16)p2.x; a[kc][9]=(_Float16)p2.y; a[kc][10]=(_Float16)p2.z; a[kc][11]=(_Float16)p2.w;
    a[kc][12]=(_Float16)p3.x; a[kc][13]=(_Float16)p3.y; a[kc][14]=(_Float16)p3.z; a[kc][15]=(_Float16)p3.w;
  }

  float rsv[8];
  if (MODE >= 2){
    #pragma unroll
    for (int r = 0; r < 8; ++r) rsv[r] = rowscale[rowBase + (hl ? r+8 : r)];
  }

  #pragma unroll
  for (int nt = 0; nt < 8; ++nt){
    const int n = nt*16 + m;
    v8f acc = {};
    #pragma unroll
    for (int kc = 0; kc < 4; ++kc){
      // B frag (32x16 f16 layout): column n, K = kc*32 + {kbB..kbB+7, kbB+16..kbB+23} in Wh row n
      const v8h* bp = (const v8h*)(Wh + n*136 + kc*32 + kbB);
      v8h lo = bp[0], hi = bp[2];
      v16h b;
      #pragma unroll
      for (int j = 0; j < 8; ++j){ b[j] = lo[j]; b[j+8] = hi[j]; }
      acc = __builtin_amdgcn_wmma_f32_16x16x32_f16(false, a[kc], false, b,
                                                   (short)0, acc, false, false);
    }
    const float bv = bsh[n];
    #pragma unroll
    for (int r = 0; r < 8; ++r){
      const int er = rowBase + (hl ? r+8 : r);
      const size_t o = (size_t)er * NF + n;
      float t = acc[r] + bv;
      float res;
      if      (MODE == 0) res = leaky(t + addend[o]);
      else if (MODE == 1) res = t;
      else if (MODE == 2) res = leaky(t * rsv[r]);
      else                res = addend[o] + leaky(t * rsv[r]);
      out[o] = res;
    }
  }
}

// ---------------- driver ----------------

extern "C" void kernel_launch(void* const* d_in, const int* in_sizes, int n_in,
                              void* d_out, int out_size, void* d_ws, size_t ws_size,
                              hipStream_t stream)
{
  const float* v    = (const float*)d_in[0];
  const float* w    = (const float*)d_in[1];
  const int*   conn = (const int*)  d_in[2];
  const float* W11  = (const float*)d_in[3];
  const float* W12  = (const float*)d_in[4];
  const float* b12  = (const float*)d_in[5];
  const float* W13  = (const float*)d_in[6];
  const float* b13  = (const float*)d_in[7];
  const float* W14  = (const float*)d_in[8];
  const float* b14  = (const float*)d_in[9];
  const float* W15  = (const float*)d_in[10];
  const float* b15  = (const float*)d_in[11];
  const float* W16  = (const float*)d_in[12];
  const float* b16  = (const float*)d_in[13];
  const float* W21  = (const float*)d_in[14];

  const int Nn    = in_sizes[0] / 5;     // 4096
  const int E     = in_sizes[1] / 4;     // 16384 (multiple of 128)
  const int words = Nn >> 5;             // bitset words per node row
  const size_t EN   = (size_t)E  * NF;
  const size_t NnNF = (size_t)Nn * NF;

  // workspace layout
  float* mu = (float*)d_ws;
  float* Ab = mu + EN;
  float* Bb = Ab + EN;
  float* Cb = Bb + EN;
  unsigned long long* acc64 = (unsigned long long*)(Cb + EN);
  unsigned* adjbits = (unsigned*)(acc64 + NnNF);
  int*   deg    = (int*)(adjbits + (size_t)Nn * words);
  float* inv1   = (float*)(deg + Nn);
  float* inv2   = inv1 + E;
  float* colsum = inv2 + E;
  float* qconst = colsum + NF;
  float* hv     = Cb;                    // alias: Cb unused until loop body

  auto cdiv = [](long long a, long long b){ return (int)((a + b - 1) / b); };

  const int adjW = Nn * words;
  k_zero_u32 <<<cdiv(adjW,256), 256, 0, stream>>>(adjbits, adjW);
  k_adj_bits <<<cdiv(E,256),    256, 0, stream>>>(conn, adjbits, E, words);
  k_degree   <<<cdiv(Nn,256),   256, 0, stream>>>(adjbits, deg, Nn, words);
  k_edge_norms<<<cdiv(E,256),   256, 0, stream>>>(conn, deg, inv1, inv2, E);
  k_hv       <<<cdiv((long long)NnNF,256), 256, 0, stream>>>(v, W13, b13, hv, (int)NnNF);
  k_h1       <<<cdiv((long long)EN,256),   256, 0, stream>>>(w, W11, Bb, (int)EN);
  k_gatherX0 <<<cdiv((long long)EN,256),   256, 0, stream>>>(conn, hv, Ab, (int)EN);

  // mu = leaky(X0 @ W12.T + b12 + h1)
  k_gemm128_wmma<0><<<E/128, 256, 0, stream>>>(Ab, W12, b12, Bb, nullptr, mu);

  for (int it = 0; it < 2; ++it){
    // h3 = mu @ W16.T + b16
    k_gemm128_wmma<1><<<E/128, 256, 0, stream>>>(mu, W16, b16, nullptr, nullptr, Ab);
    // h4_0 = IA @ mu (deterministic fixed-point scatter)
    k_zero_u32 <<<cdiv((long long)(NnNF*2),256), 256, 0, stream>>>((unsigned*)acc64, (int)(NnNF*2));
    k_scatter  <<<cdiv((long long)EN,256), 256, 0, stream>>>(mu, conn, acc64, (int)EN);
    k_make_G   <<<cdiv((long long)EN,256), 256, 0, stream>>>(mu, conn, acc64, Bb, Cb, (int)EN);
    // Bb = leaky((G1 @ W14.T + b14) / n1)          (in-place: each wave reads its own rows first)
    k_gemm128_wmma<2><<<E/128, 256, 0, stream>>>(Bb, W14, b14, nullptr, inv1, Bb);
    // Bb += leaky((G2 @ W14.T + b14) / n2)
    k_gemm128_wmma<3><<<E/128, 256, 0, stream>>>(Cb, W14, b14, Bb, inv2, Bb);
    // mu = leaky(Bb @ W15.T + b15 + h3)
    k_gemm128_wmma<0><<<E/128, 256, 0, stream>>>(Bb, W15, b15, Ab, nullptr, mu);
  }

  k_colsum<<<NF, 256, 0, stream>>>(mu, colsum, E);
  k_qconst<<<1, 1, 0, stream>>>(colsum, W21, qconst);
  k_final <<<cdiv(E,256), 256, 0, stream>>>(mu, W21, qconst, (float*)d_out, E);
}
